// Block_10359461118546
// MI455X (gfx1250) — matrix operand order repaired and checked by reading, not compiled, not run on hardware
//
#include <hip/hip_runtime.h>
#include <hip/hip_bf16.h>
#include <math.h>

// ---------------- problem constants ----------------
#define BB 32
#define NN 2048
#define DD 512
#define DGG 1024
#define BN (BB * NN) // 65536 tokens

typedef __bf16 bf16_t;
typedef bf16_t v16bf __attribute__((ext_vector_type(16)));
typedef bf16_t v8bf  __attribute__((ext_vector_type(8)));
typedef float  v8f   __attribute__((ext_vector_type(8)));
typedef int    v4i_g __attribute__((vector_size(16))); // matches builtin param type

enum { EPI_BIAS = 0, EPI_SILU = 1, EPI_SIGMOID = 2 };

#define AS1 __attribute__((address_space(1)))
#define AS3 __attribute__((address_space(3)))

#if defined(__has_builtin)
#if __has_builtin(__builtin_amdgcn_global_load_async_to_lds_b128)
#define USE_ASYNC_LDS 1
#endif
#endif

// 16B global->LDS copy: async (ASYNCcnt, bypasses VGPRs) when available.
__device__ __forceinline__ void cp16_g2s(const bf16_t* g, bf16_t* l) {
#ifdef USE_ASYNC_LDS
  __builtin_amdgcn_global_load_async_to_lds_b128(
      (AS1 v4i_g*)(uintptr_t)g, (AS3 v4i_g*)(uintptr_t)l, 0, 0);
#else
  *(uint4*)l = *(const uint4*)g;
#endif
}

__device__ __forceinline__ void cp_wait() {
#ifdef USE_ASYNC_LDS
#if __has_builtin(__builtin_amdgcn_s_wait_asynccnt)
  __builtin_amdgcn_s_wait_asynccnt(0);
#else
  asm volatile("s_wait_asynccnt 0" ::: "memory");
#endif
#endif
}

__device__ __forceinline__ v16bf cat8(v8bf lo, v8bf hi) {
  v16bf r;
#pragma unroll
  for (int i = 0; i < 8; ++i) { r[i] = lo[i]; r[i + 8] = hi[i]; }
  return r;
}

__device__ __forceinline__ float sigmoidf_(float z) { return 1.0f / (1.0f + __expf(-z)); }

// ---------------- bf16 WMMA GEMM: C[M,N] = act(A[M,K] @ Wt[N,K]^T + bias) ----------------
// A row-major bf16; Wt row-major bf16 with shape [N,K] (W pre-transposed).
// Block tile 128x128, 8 waves (wave32), wave tile 32(M) x 64(N), K step 32.
// Double-buffered LDS; async global->LDS staging overlapped with WMMA.
template <typename OutT, int EPI>
__global__ __launch_bounds__(256) void gemm_bf16_wmma(
    const bf16_t* __restrict__ A, const bf16_t* __restrict__ Wt,
    const float* __restrict__ bias, OutT* __restrict__ C, int Ncols, int K) {
  __shared__ __align__(16) bf16_t As[2][128 * 40]; // 32 halfs/row + 8 pad
  __shared__ __align__(16) bf16_t Bs[2][128 * 40];

  const int tid  = threadIdx.x;
  const int wid  = tid >> 5;
  const int lane = tid & 31;
  const int wm   = wid & 3;  // 4 waves along M
  const int wn   = wid >> 2; // 2 waves along N
  const int bm   = blockIdx.y * 128;
  const int bn   = blockIdx.x * 128;
  const int half = lane >> 4;
  const int l16  = lane & 15;

  v8f acc[2][4] = {};

  const int ksteps = K >> 5;

  // stage one 128x32 tile pair (A,Wt): 512 x 16B segments each, 256 threads
  auto stage = [&](int buf, int ks) {
    const int k0 = ks << 5;
#pragma unroll
    for (int i = 0; i < 2; ++i) {
      const int id  = tid + i * 256;
      const int row = id >> 2;
      const int seg = id & 3;
      cp16_g2s(A + (size_t)(bm + row) * K + k0 + seg * 8, &As[buf][row * 40 + seg * 8]);
      cp16_g2s(Wt + (size_t)(bn + row) * K + k0 + seg * 8, &Bs[buf][row * 40 + seg * 8]);
    }
  };

  stage(0, 0);
  for (int ks = 0; ks < ksteps; ++ks) {
    const int buf = ks & 1;
    cp_wait();        // my async writes into buf are complete
    __syncthreads();  // everyone's writes done; everyone done reading buf^1
    if (ks + 1 < ksteps) stage(buf ^ 1, ks + 1); // overlap next tile with compute

    // A fragments: row of wave tile, two 8-half chunks at k = 8*half and +16
    v16bf af[2];
#pragma unroll
    for (int mt = 0; mt < 2; ++mt) {
      const int r    = wm * 32 + mt * 16 + l16;
      const int base = r * 40 + half * 8;
      af[mt] = cat8(*(const v8bf*)&As[buf][base], *(const v8bf*)&As[buf][base + 16]);
    }
    // B fragments: row n of Wt tile, two 8-half chunks at k = 8*half and +16
    v16bf bfr[4];
#pragma unroll
    for (int nt = 0; nt < 4; ++nt) {
      const int r    = wn * 64 + nt * 16 + l16;
      const int base = r * 40 + half * 8;
      bfr[nt] = cat8(*(const v8bf*)&Bs[buf][base], *(const v8bf*)&Bs[buf][base + 16]);
    }
#pragma unroll
    for (int mt = 0; mt < 2; ++mt)
#pragma unroll
      for (int nt = 0; nt < 4; ++nt)
        acc[mt][nt] = __builtin_amdgcn_wmma_f32_16x16x32_bf16(
            false, af[mt], false, bfr[nt], (short)0, acc[mt][nt], false, false);
  }

  // epilogue: D layout VGPR r -> M = r + 8*half, N = l16
#pragma unroll
  for (int nt = 0; nt < 4; ++nt) {
    const int col = bn + wn * 64 + nt * 16 + l16;
    const float bv = bias[col];
#pragma unroll
    for (int mt = 0; mt < 2; ++mt) {
#pragma unroll
      for (int r = 0; r < 8; ++r) {
        const int row = bm + wm * 32 + mt * 16 + half * 8 + r;
        float z = acc[mt][nt][r] + bv;
        if (EPI == EPI_SILU)    z = z * sigmoidf_(z);
        if (EPI == EPI_SIGMOID) z = sigmoidf_(z);
        C[(size_t)row * Ncols + col] = (OutT)z;
      }
    }
  }
}

// ---------------- weight convert + transpose: W[K,N] f32 -> Wt[N,K] bf16 ----------------
__global__ void wt_convert(const float* __restrict__ W, bf16_t* __restrict__ Wt, int K, int N) {
  const size_t tot = (size_t)K * N;
  for (size_t i = blockIdx.x * blockDim.x + threadIdx.x; i < tot; i += (size_t)gridDim.x * blockDim.x) {
    const size_t k = i / N, n = i - k * N;
    Wt[n * K + k] = (bf16_t)W[i];
  }
}

__global__ void f32_to_bf16(const float* __restrict__ in, bf16_t* __restrict__ out, size_t tot) {
  for (size_t i = blockIdx.x * blockDim.x + threadIdx.x; i < tot; i += (size_t)gridDim.x * blockDim.x)
    out[i] = (bf16_t)in[i];
}

// ---------------- gates: i(F0),f(F1) -> u(F0), lam(F1) in place ----------------
__global__ void gates_kernel(float* __restrict__ iu, float* __restrict__ flam,
                             const int* __restrict__ lb_ptr) {
  const float lb = (float)lb_ptr[0];
  const size_t tot = (size_t)BN * DD;
  for (size_t i = blockIdx.x * blockDim.x + threadIdx.x; i < tot; i += (size_t)gridDim.x * blockDim.x) {
    const float f   = flam[i];
    const float lam = lb + (1.0f - lb) * f;
    const float u   = (1.0f - lam) * iu[i];
    flam[i] = lam;
    iu[i]   = u;
  }
}

// ---------------- bidirectional linear recurrence along N ----------------
__global__ __launch_bounds__(256) void hgru_scan(const float* __restrict__ lam,
                                                 const float* __restrict__ u,
                                                 float* __restrict__ h) {
  const int gid = blockIdx.x * blockDim.x + threadIdx.x; // one (b,d) chain per lane
  if (gid >= BB * DD) return;
  const int b = gid / DD, d = gid - b * DD;
  const size_t base = (size_t)b * NN * DD + d;
  float s = 0.0f;
  for (int n = 0; n < NN; ++n) {
    const size_t i = base + (size_t)n * DD;
    s = lam[i] * s + u[i];
    h[i] = s;
  }
  s = 0.0f;
  for (int n = NN - 1; n >= 0; --n) {
    const size_t i = base + (size_t)n * DD;
    s = lam[i] * s + u[i];
    h[i] += s;
  }
}

__device__ __forceinline__ float block_reduce_sum(float v, float* red) {
  const int tid = threadIdx.x;
  red[tid] = v;
  __syncthreads();
  for (int s = 128; s > 0; s >>= 1) {
    if (tid < s) red[tid] += red[tid + s];
    __syncthreads();
  }
  const float r = red[0];
  __syncthreads();
  return r;
}

// ---------------- hn = simple_rms_norm(h) * g -> bf16 ----------------
__global__ __launch_bounds__(256) void hnorm_gate(const float* __restrict__ h,
                                                  const float* __restrict__ g,
                                                  bf16_t* __restrict__ out) {
  __shared__ float red[256];
  const int tid = threadIdx.x;
  const size_t base = (size_t)blockIdx.x * DD;
  const float v0 = h[base + tid], v1 = h[base + tid + 256];
  const float ss = block_reduce_sum(v0 * v0 + v1 * v1, red);
  const float rms = rsqrtf(ss / (float)DD + 1e-6f);
  out[base + tid]       = (bf16_t)(v0 * rms * g[base + tid]);
  out[base + tid + 256] = (bf16_t)(v1 * rms * g[base + tid + 256]);
}

// ---------------- out = xin + layernorm(t)*gamma+beta ; optional bf16 copy ----------------
template <bool WB16>
__global__ __launch_bounds__(256) void resid_ln(const float* __restrict__ xin,
                                                const float* __restrict__ t,
                                                const float* __restrict__ gamma,
                                                const float* __restrict__ beta,
                                                float* __restrict__ outf,
                                                bf16_t* __restrict__ outb) {
  __shared__ float red[256];
  const int tid = threadIdx.x;
  const size_t base = (size_t)blockIdx.x * DD;
  const float t0 = t[base + tid], t1 = t[base + tid + 256];
  const float mean = block_reduce_sum(t0 + t1, red) / (float)DD;
  const float msq  = block_reduce_sum(t0 * t0 + t1 * t1, red) / (float)DD;
  const float inv  = rsqrtf(msq - mean * mean + 1e-5f);
  {
    const int d = tid;
    const float o = xin[base + d] + (t0 - mean) * inv * gamma[d] + beta[d];
    outf[base + d] = o;
    if (WB16) outb[base + d] = (bf16_t)o;
  }
  {
    const int d = tid + 256;
    const float o = xin[base + d] + (t1 - mean) * inv * gamma[d] + beta[d];
    outf[base + d] = o;
    if (WB16) outb[base + d] = (bf16_t)o;
  }
}

// ---------------- v = z1 * z2 (bf16, z1 already silu'd) ----------------
__global__ void mul_bf16(bf16_t* __restrict__ z1, const bf16_t* __restrict__ z2, size_t tot) {
  for (size_t i = blockIdx.x * blockDim.x + threadIdx.x; i < tot; i += (size_t)gridDim.x * blockDim.x)
    z1[i] = (bf16_t)((float)z1[i] * (float)z2[i]);
}

// ---------------- launch ----------------
extern "C" void kernel_launch(void* const* d_in, const int* in_sizes, int n_in,
                              void* d_out, int out_size, void* d_ws, size_t ws_size,
                              hipStream_t stream) {
  (void)in_sizes; (void)n_in; (void)out_size; (void)ws_size;
  const float* x   = (const float*)d_in[0];
  const int*   lb  = (const int*)d_in[1];
  const float* Wi  = (const float*)d_in[2];  const float* bi = (const float*)d_in[3];
  const float* Wf  = (const float*)d_in[4];  const float* bf = (const float*)d_in[5];
  const float* Wg  = (const float*)d_in[6];  const float* bg = (const float*)d_in[7];
  const float* Wo  = (const float*)d_in[8];  const float* bo = (const float*)d_in[9];
  const float* tng = (const float*)d_in[10]; const float* tnb = (const float*)d_in[11];
  const float* fng = (const float*)d_in[12]; const float* fnb = (const float*)d_in[13];
  const float* W1  = (const float*)d_in[14]; const float* b1 = (const float*)d_in[15];
  const float* W2  = (const float*)d_in[16]; const float* b2 = (const float*)d_in[17];
  const float* W3  = (const float*)d_in[18]; const float* b3 = (const float*)d_in[19];
  float* out = (float*)d_out;

  // workspace arena
  char* w = (char*)d_ws;
  size_t off = 0;
  auto carve = [&](size_t bytes) { char* p = w + off; off += (bytes + 255) & ~(size_t)255; return p; };
  bf16_t* WtI = (bf16_t*)carve((size_t)DD * DD * 2);
  bf16_t* WtF = (bf16_t*)carve((size_t)DD * DD * 2);
  bf16_t* WtG = (bf16_t*)carve((size_t)DD * DD * 2);
  bf16_t* WtO = (bf16_t*)carve((size_t)DD * DD * 2);
  bf16_t* Wt1 = (bf16_t*)carve((size_t)DD * DGG * 2); // [DGG, DD]
  bf16_t* Wt2 = (bf16_t*)carve((size_t)DD * DGG * 2); // [DGG, DD]
  bf16_t* Wt3 = (bf16_t*)carve((size_t)DGG * DD * 2); // [DD, DGG]
  bf16_t* xb  = (bf16_t*)carve((size_t)BN * DD * 2);  // x in bf16; later x1 in bf16
  bf16_t* hb  = (bf16_t*)carve((size_t)BN * DD * 2);  // rmsnorm(h)*g in bf16
  bf16_t* z1b = (bf16_t*)carve((size_t)BN * DGG * 2); // silu(xW1+b1), then v
  bf16_t* z2b = (bf16_t*)carve((size_t)BN * DGG * 2); // xW2+b2
  float*  F0  = (float*)carve((size_t)BN * DD * 4);   // i -> u -> g -> t2
  float*  F1  = (float*)carve((size_t)BN * DD * 4);   // f -> lam -> t(Wo)
  float*  F2  = (float*)carve((size_t)BN * DD * 4);   // h -> x1

  const dim3 tpb(256);
  const dim3 gemmD(DD / 128, BN / 128);   // (4, 512)
  const dim3 gemmG(DGG / 128, BN / 128);  // (8, 512)
  const dim3 rows(BN);                    // one block per token

  // 0) weight conversion (tiny)
  wt_convert<<<512, tpb, 0, stream>>>(Wi, WtI, DD, DD);
  wt_convert<<<512, tpb, 0, stream>>>(Wf, WtF, DD, DD);
  wt_convert<<<512, tpb, 0, stream>>>(Wg, WtG, DD, DD);
  wt_convert<<<512, tpb, 0, stream>>>(Wo, WtO, DD, DD);
  wt_convert<<<512, tpb, 0, stream>>>(W1, Wt1, DD, DGG);
  wt_convert<<<512, tpb, 0, stream>>>(W2, Wt2, DD, DGG);
  wt_convert<<<512, tpb, 0, stream>>>(W3, Wt3, DGG, DD);
  // 1) x -> bf16
  f32_to_bf16<<<8192, tpb, 0, stream>>>(x, xb, (size_t)BN * DD);
  // 2) i = silu(x@Wi + bi) ; f = sigmoid(x@Wf + bf)
  gemm_bf16_wmma<float, EPI_SILU><<<gemmD, tpb, 0, stream>>>(xb, WtI, bi, F0, DD, DD);
  gemm_bf16_wmma<float, EPI_SIGMOID><<<gemmD, tpb, 0, stream>>>(xb, WtF, bf, F1, DD, DD);
  // 3) lam = lb + (1-lb)*f ; u = (1-lam)*i   (in place)
  gates_kernel<<<8192, tpb, 0, stream>>>(F0, F1, lb);
  // 4) h = scan_fwd + scan_bwd
  hgru_scan<<<(BB * DD + 255) / 256, tpb, 0, stream>>>(F1, F0, F2);
  // 5) g = sigmoid(x@Wg + bg)
  gemm_bf16_wmma<float, EPI_SIGMOID><<<gemmD, tpb, 0, stream>>>(xb, WtG, bg, F0, DD, DD);
  // 6) hb = rmsnorm(h) * g   (bf16)
  hnorm_gate<<<rows, tpb, 0, stream>>>(F2, F0, hb);
  // 7) t = hb@Wo + bo
  gemm_bf16_wmma<float, EPI_BIAS><<<gemmD, tpb, 0, stream>>>(hb, WtO, bo, F1, DD, DD);
  // 8) x1 = x + LN(t; tn)  -> F2 (f32) + xb (bf16, reuse)
  resid_ln<true><<<rows, tpb, 0, stream>>>(x, F1, tng, tnb, F2, xb);
  // 9) z1 = silu(x1@W1 + b1) (bf16) ; z2 = x1@W2 + b2 (bf16)
  gemm_bf16_wmma<bf16_t, EPI_SILU><<<gemmG, tpb, 0, stream>>>(xb, Wt1, b1, z1b, DGG, DD);
  gemm_bf16_wmma<bf16_t, EPI_BIAS><<<gemmG, tpb, 0, stream>>>(xb, Wt2, b2, z2b, DGG, DD);
  // 10) v = z1 * z2 (in place into z1b)
  mul_bf16<<<16384, tpb, 0, stream>>>(z1b, z2b, (size_t)BN * DGG);
  // 11) t2 = v@W3 + b3
  gemm_bf16_wmma<float, EPI_BIAS><<<gemmD, tpb, 0, stream>>>(z1b, Wt3, b3, F0, DD, DGG);
  // 12) out = x1 + LN(t2; fn)
  resid_ln<false><<<rows, tpb, 0, stream>>>(F2, F0, fng, fnb, out, nullptr);
}
